// SOONet_51582557225161
// MI455X (gfx1250) — compile-verified, matrix operand order checked by reading, not run on hardware
//
#include <hip/hip_runtime.h>
#include <hip/hip_bf16.h>
#include <math.h>

typedef __attribute__((ext_vector_type(16))) _Float16 v16h;
typedef __attribute__((ext_vector_type(8)))  _Float16 v8h;
typedef __attribute__((ext_vector_type(8)))  float    v8f;

#define T_FRAMES 100000
#define D_DIM    512
#define Q_DIM    128
#define H_DIM    512
#define E_NUM    1000
#define E_PAD    1008      // 63 tiles of 16
#define TOPK     100
#define FRATE    5.0f

// ---------------- segment reduction ----------------

__global__ void k_init_events(int* __restrict__ start, int* __restrict__ end,
                              int* __restrict__ cnt) {
  int i = blockIdx.x * blockDim.x + threadIdx.x;
  if (i < E_NUM) { start[i] = T_FRAMES; end[i] = 0; cnt[i] = 0; }
}

__global__ void k_seg_minmax(const int* __restrict__ ids, int* __restrict__ start,
                             int* __restrict__ end, int* __restrict__ cnt) {
  int t = blockIdx.x * blockDim.x + threadIdx.x;
  if (t < T_FRAMES) {
    int e = ids[t];
    atomicMin(&start[e], t);
    atomicMax(&end[e], t);
    atomicAdd(&cnt[e], 1);
  }
}

// one block per event; ids sorted => event frames are contiguous [s, en]
__global__ void k_event_avg(const float* __restrict__ feats,
                            const int* __restrict__ start, const int* __restrict__ end,
                            const int* __restrict__ cnt,
                            float* __restrict__ avg, _Float16* __restrict__ Ae) {
  int e = blockIdx.x;
  int s = start[e], en = end[e], c = cnt[e];
  float inv = (c > 0) ? 1.0f / (float)c : 0.0f;
  for (int d = threadIdx.x; d < D_DIM; d += blockDim.x) {
    float acc = 0.0f;
    if (c > 0) {
      for (int t = s; t <= en; ++t) acc += feats[(size_t)t * D_DIM + d];
    }
    float a = acc * inv;
    avg[(size_t)e * D_DIM + d] = a;
    Ae[(size_t)e * D_DIM + d] = (_Float16)a;
  }
}

__global__ void k_pad_Ae(_Float16* __restrict__ Ae) {
  int i = blockIdx.x * blockDim.x + threadIdx.x;      // (E_PAD-E_NUM)*D_DIM = 4096
  if (i < (E_PAD - E_NUM) * D_DIM) Ae[(size_t)E_NUM * D_DIM + i] = (_Float16)0.0f;
}

__global__ void k_f32_to_f16(const float* __restrict__ src, _Float16* __restrict__ dst,
                             int n) {
  int i = blockIdx.x * blockDim.x + threadIdx.x;
  if (i < n) dst[i] = (_Float16)src[i];
}

// W[k][n] (512x512 f32, row major) -> WT[n][k] f16 (B^T so B-fragment loads mirror A)
__global__ void k_transpose_cvt(const float* __restrict__ W, _Float16* __restrict__ WT) {
  int i = blockIdx.x * blockDim.x + threadIdx.x;      // 512*512
  int n = i & 511, k = i >> 9;                        // coalesced read of W
  WT[(size_t)n * 512 + k] = (_Float16)W[i];
}

// ---------------- WMMA GEMM: C[M,N] = A[M,K] * B[K,N], BT = B^T, K = 512 ----------------
// One wave per 16x16 output tile. 8 waves / block. Grids are exact (EXEC all-1s).

__global__ void k_wmma_gemm(const _Float16* __restrict__ A, const _Float16* __restrict__ BT,
                            float* __restrict__ C, int mTiles, int nTiles) {
  int lane = threadIdx.x & 31;
  int wave = threadIdx.x >> 5;
  int tile = blockIdx.x * 8 + wave;
  if (tile >= mTiles * nTiles) return;                // wave-uniform
  int mT = tile / nTiles;
  int nT = tile % nTiles;

  int rsel = lane & 15;
  int kSel = (lane >> 4) << 3;                        // 0 or 8 (ISA 16-bit A layout)
  const _Float16* ap = A  + ((size_t)(mT * 16 + rsel)) * D_DIM + kSel;
  const _Float16* bp = BT + ((size_t)(nT * 16 + rsel)) * D_DIM + kSel;

  v8f acc = {};
#pragma unroll 4
  for (int k0 = 0; k0 < D_DIM; k0 += 32) {
    __builtin_prefetch(ap + k0 + 64, 0, 0);
    __builtin_prefetch(bp + k0 + 64, 0, 0);
    union { v16h v; v8h h[2]; } a, b;
    a.h[0] = *(const v8h*)(ap + k0);
    a.h[1] = *(const v8h*)(ap + k0 + 16);
    b.h[0] = *(const v8h*)(bp + k0);
    b.h[1] = *(const v8h*)(bp + k0 + 16);
    acc = __builtin_amdgcn_wmma_f32_16x16x32_f16(false, a.v, false, b.v,
                                                 (short)0, acc, false, false);
  }

  int m0 = mT * 16 + ((lane >> 4) << 3);
  int n  = nT * 16 + rsel;
#pragma unroll
  for (int i = 0; i < 8; ++i)
    C[(size_t)(m0 + i) * (size_t)(nTiles * 16) + n] = acc[i];
}

// ---------------- regressor head: pred_timestamps [E,2] ----------------

__global__ void k_pred(const float* __restrict__ avg, const int* __restrict__ start,
                       const int* __restrict__ end, const float* __restrict__ Wr,
                       const float* __restrict__ br, float* __restrict__ pred) {
  __shared__ float s0[256], s1[256];
  int e = blockIdx.x, tid = threadIdx.x;
  float p0 = 0.0f, p1 = 0.0f;
  for (int d = tid; d < D_DIM; d += 256) {
    float a = avg[(size_t)e * D_DIM + d];
    p0 += a * Wr[d * 2 + 0];
    p1 += a * Wr[d * 2 + 1];
  }
  s0[tid] = p0; s1[tid] = p1;
  __syncthreads();
  for (int o = 128; o > 0; o >>= 1) {
    if (tid < o) { s0[tid] += s0[tid + o]; s1[tid] += s1[tid + o]; }
    __syncthreads();
  }
  if (tid == 0) {
    float st = (float)start[e], en = (float)end[e];
    float dur = (en - st + 1.0f) / FRATE;
    float bb0 = tanhf(s0[0] + br[0]);
    float bb1 = tanhf(s1[0] + br[1]);
    pred[e * 2 + 0] = st / FRATE + bb0 * dur;
    pred[e * 2 + 1] = en / FRATE + bb1 * dur;
  }
}

// ---------------- fused scores: sigmoid( relu(qh+eh+b1) @ W2 + b2 ) ----------------
// block = (q, group of 8 events); wave per event; qh[q]+b1 and W2 cached in LDS.

__global__ void k_scores(const float* __restrict__ qh, const float* __restrict__ eh,
                         const float* __restrict__ b1, const float* __restrict__ W2,
                         const float* __restrict__ b2, float* __restrict__ scores) {
  __shared__ float qb[H_DIM];
  __shared__ float w2s[H_DIM];
  int q = blockIdx.y, tid = threadIdx.x;
  for (int h = tid; h < H_DIM; h += 256) {
    qb[h]  = qh[(size_t)q * H_DIM + h] + b1[h];
    w2s[h] = W2[h];
  }
  __syncthreads();
  int lane = tid & 31, wave = tid >> 5;
  int e = blockIdx.x * 8 + wave;
  if (e < E_NUM) {
    const float* ep = eh + (size_t)e * H_DIM;
    float sum = 0.0f;
#pragma unroll 4
    for (int h = lane; h < H_DIM; h += 32) {
      float v = qb[h] + ep[h];
      v = v > 0.0f ? v : 0.0f;
      sum += v * w2s[h];
    }
    for (int o = 16; o > 0; o >>= 1) sum += __shfl_xor(sum, o, 32);
    if (lane == 0)
      scores[(size_t)q * E_NUM + e] = 1.0f / (1.0f + expf(-(sum + b2[0])));
  }
}

// ---------------- top-100 per query (iterative argmax, first-index tie-break) ----------------

__global__ void k_topk(const float* __restrict__ scores, float* __restrict__ outIdx) {
  __shared__ float vals[E_NUM];
  __shared__ float rv[256];
  __shared__ int   ri[256];
  int q = blockIdx.x, tid = threadIdx.x;
  for (int e = tid; e < E_NUM; e += 256) vals[e] = scores[(size_t)q * E_NUM + e];
  __syncthreads();
  for (int k = 0; k < TOPK; ++k) {
    float bv = -INFINITY; int bi = 0x7fffffff;
    for (int e = tid; e < E_NUM; e += 256) {
      float v = vals[e];
      if (v > bv || (v == bv && e < bi)) { bv = v; bi = e; }
    }
    rv[tid] = bv; ri[tid] = bi;
    __syncthreads();
    for (int o = 128; o > 0; o >>= 1) {
      if (tid < o) {
        float ov = rv[tid + o]; int oi = ri[tid + o];
        if (ov > rv[tid] || (ov == rv[tid] && oi < ri[tid])) { rv[tid] = ov; ri[tid] = oi; }
      }
      __syncthreads();
    }
    if (tid == 0) { outIdx[(size_t)q * TOPK + k] = (float)ri[0]; vals[ri[0]] = -INFINITY; }
    __syncthreads();
  }
}

// ---------------- launch ----------------

extern "C" void kernel_launch(void* const* d_in, const int* in_sizes, int n_in,
                              void* d_out, int out_size, void* d_ws, size_t ws_size,
                              hipStream_t stream) {
  const float* query_feats = (const float*)d_in[0];   // [128,512]
  const float* video_feats = (const float*)d_in[1];   // [1,100000,512]
  const float* W1q         = (const float*)d_in[2];   // [512,512]
  const float* W1e         = (const float*)d_in[3];   // [512,512]
  const float* b1          = (const float*)d_in[4];   // [512]
  const float* W2          = (const float*)d_in[5];   // [512,1]
  const float* b2          = (const float*)d_in[6];   // [1]
  const float* Wr          = (const float*)d_in[7];   // [512,2]
  const float* br          = (const float*)d_in[8];   // [2]
  const int*   ids         = (const int*)d_in[9];     // [100000]

  char* ws = (char*)d_ws;
  int*      w_start = (int*)(ws + 0);                   // 1024 ints
  int*      w_end   = (int*)(ws + 4096);
  int*      w_cnt   = (int*)(ws + 8192);
  float*    w_avg   = (float*)(ws + 12288);             // 1000*512 f32
  _Float16* w_Ae    = (_Float16*)(ws + 2060288);        // 1008*512 f16
  _Float16* w_Aq    = (_Float16*)(ws + 3092480);        // 128*512  f16
  _Float16* w_WeT   = (_Float16*)(ws + 3223552);        // 512*512  f16
  _Float16* w_WqT   = (_Float16*)(ws + 3747840);        // 512*512  f16
  float*    w_eh    = (float*)(ws + 4272128);           // 1008*512 f32
  float*    w_qh    = (float*)(ws + 6336512);           // 128*512  f32

  float* out_scores = (float*)d_out;                    // [128*1000]
  float* out_pred   = out_scores + Q_DIM * E_NUM;       // [1000*2]
  float* out_topidx = out_pred + E_NUM * 2;             // [128*100]

  k_init_events<<<4, 256, 0, stream>>>(w_start, w_end, w_cnt);
  k_seg_minmax<<<(T_FRAMES + 255) / 256, 256, 0, stream>>>(ids, w_start, w_end, w_cnt);
  k_event_avg<<<E_NUM, 256, 0, stream>>>(video_feats, w_start, w_end, w_cnt, w_avg, w_Ae);
  k_pad_Ae<<<16, 256, 0, stream>>>(w_Ae);
  k_f32_to_f16<<<(Q_DIM * D_DIM + 255) / 256, 256, 0, stream>>>(query_feats, w_Aq, Q_DIM * D_DIM);
  k_transpose_cvt<<<1024, 256, 0, stream>>>(W1e, w_WeT);
  k_transpose_cvt<<<1024, 256, 0, stream>>>(W1q, w_WqT);

  // eh = avg @ W1e : 63x32 tiles = 2016 = 252 blocks * 8 waves (exact)
  k_wmma_gemm<<<252, 256, 0, stream>>>(w_Ae, w_WeT, w_eh, E_PAD / 16, H_DIM / 16);
  // qh = query @ W1q : 8x32 tiles = 256 = 32 blocks * 8 waves (exact)
  k_wmma_gemm<<<32, 256, 0, stream>>>(w_Aq, w_WqT, w_qh, Q_DIM / 16, H_DIM / 16);

  k_pred<<<E_NUM, 256, 0, stream>>>(w_avg, w_start, w_end, Wr, br, out_pred);
  k_scores<<<dim3(E_NUM / 8, Q_DIM), 256, 0, stream>>>(w_qh, w_eh, b1, W2, b2, out_scores);
  k_topk<<<Q_DIM, 256, 0, stream>>>(out_scores, out_topidx);
}
